// HyperSelfAttention_63050119905324
// MI455X (gfx1250) — compile-verified, matrix operand order checked by reading, not run on hardware
//
#include <hip/hip_runtime.h>
#include <hip/hip_bf16.h>
#include <math.h>

typedef __bf16 bf16;
typedef __bf16 bf16x8  __attribute__((ext_vector_type(8)));
typedef __bf16 bf16x16 __attribute__((ext_vector_type(16)));
typedef float  f32x8   __attribute__((ext_vector_type(8)));

#define NB 2
#define SS 2048
#define DD 1024
#define NH 16
#define HD 64
#define MM (NB*SS)      // 4096 rows
#define HID 256
#define DYN 64
#define RK 8

static __device__ __forceinline__ bf16x16 mk16(bf16x8 lo, bf16x8 hi) {
    bf16x16 r;
#pragma unroll
    for (int i = 0; i < 8; ++i) { r[i] = lo[i]; r[i + 8] = hi[i]; }
    return r;
}

// ---------------------------------------------------------------- converts
__global__ __launch_bounds__(256) void cvt_bf16_kernel(const float* __restrict__ in,
                                                       bf16* __restrict__ out, int n) {
    int i = blockIdx.x * 256 + threadIdx.x;
    if (i < n) out[i] = (bf16)in[i];
}

// Wt[n][k] = (bf16)W[k][n], D x D, LDS-tiled
__global__ __launch_bounds__(256) void wtrans_kernel(const float* __restrict__ W,
                                                     bf16* __restrict__ Wt) {
    __shared__ float t[32][33];
    const int bx = blockIdx.x * 32, by = blockIdx.y * 32;
    const int tx = threadIdx.x & 31, ty = threadIdx.x >> 5;  // 32x8
#pragma unroll
    for (int r = 0; r < 32; r += 8)
        t[ty + r][tx] = W[(size_t)(by + ty + r) * DD + bx + tx];
    __syncthreads();
#pragma unroll
    for (int r = 0; r < 32; r += 8)
        Wt[(size_t)(bx + ty + r) * DD + by + tx] = (bf16)t[tx][ty + r];
}

// ---------------------------------------------------------------- hyper-net (tiny, VALU)
__global__ __launch_bounds__(256) void mean_kernel(const float* __restrict__ x,
                                                   float* __restrict__ z) {
    int idx = blockIdx.x * 256 + threadIdx.x;       // B*D
    if (idx >= NB * DD) return;
    int d = idx & (DD - 1), b = idx >> 10;
    const float* p = x + (size_t)b * SS * DD + d;
    float acc = 0.f;
    for (int s = 0; s < SS; ++s) acc += p[(size_t)s * DD];
    z[idx] = acc * (1.0f / SS);
}

// out[b][j] = act(bias[j] + sum_i in[b][i]*W[i*J+j]); act 1 = silu
__global__ __launch_bounds__(256) void dense_small_kernel(const float* __restrict__ in,
                                                          const float* __restrict__ W,
                                                          const float* __restrict__ bias,
                                                          float* __restrict__ out,
                                                          int I, int J, int act) {
    int idx = blockIdx.x * 256 + threadIdx.x;
    if (idx >= NB * J) return;
    int b = idx / J, j = idx - b * J;
    const float* ib = in + (size_t)b * I;
    float acc = bias[j];
    for (int i = 0; i < I; ++i) acc += ib[i] * W[(size_t)i * J + j];
    if (act) acc = acc / (1.0f + __expf(-acc));
    out[idx] = acc;
}

// xa[row][r] = sum_d x[row][d] * A[b][r*D+d]   (block per row)
__global__ __launch_bounds__(256) void xa_kernel(const float* __restrict__ x,
                                                 const float* __restrict__ Amat,
                                                 float* __restrict__ xa) {
    const int row = blockIdx.x;                 // B*S
    const int b = row / SS;
    const float* xr = x + (size_t)row * DD;
    const float* Ab = Amat + (size_t)b * RK * DD;
    float acc[RK];
#pragma unroll
    for (int r = 0; r < RK; ++r) acc[r] = 0.f;
    for (int d = threadIdx.x; d < DD; d += 256) {
        float xv = xr[d];
#pragma unroll
        for (int r = 0; r < RK; ++r) acc[r] += xv * Ab[(size_t)r * DD + d];
    }
    __shared__ float red[RK][256];
#pragma unroll
    for (int r = 0; r < RK; ++r) red[r][threadIdx.x] = acc[r];
    __syncthreads();
    for (int off = 128; off > 0; off >>= 1) {
        if ((int)threadIdx.x < off)
#pragma unroll
            for (int r = 0; r < RK; ++r) red[r][threadIdx.x] += red[r][threadIdx.x + off];
        __syncthreads();
    }
    if (threadIdx.x < RK) xa[(size_t)row * RK + threadIdx.x] = red[threadIdx.x][0];
}

// dst[row][d] += (1/8) * sum_r xa[row][r] * Bm[b][d][r]
__global__ __launch_bounds__(256) void add_dyn_kernel(float* __restrict__ dst,
                                                      const float* __restrict__ xa,
                                                      const float* __restrict__ Bm) {
    int idx = blockIdx.x * 256 + threadIdx.x;   // B*S*D
    if (idx >= MM * DD) return;
    int d = idx & (DD - 1), row = idx >> 10, b = row / SS;
    const float* xr = xa + (size_t)row * RK;
    const float* br = Bm + ((size_t)b * DD + d) * RK;
    float acc = 0.f;
#pragma unroll
    for (int r = 0; r < RK; ++r) acc += xr[r] * br[r];
    dst[idx] += acc * (1.0f / RK);
}

// ---------------------------------------------------------------- WMMA GEMM
// C[M,N] = A[M,K]@W + bias ; W given transposed as Bt[N,K] (bf16).
// Block 256 thr = 8 waves (4 M x 2 N), block tile 128x128, wave tile 32x64
// (2x4 of 16x16 f32 frags = 8 WMMA per K-step of 32). Next-tile global_prefetch.
__global__ __launch_bounds__(256) void gemm_bf16_kernel(const bf16* __restrict__ A,
                                                        const bf16* __restrict__ Bt,
                                                        const float* __restrict__ bias,
                                                        float* __restrict__ C,
                                                        int M, int N, int K) {
    __shared__ __align__(16) bf16 sA[128][40];
    __shared__ __align__(16) bf16 sB[128][40];
    const int tid = threadIdx.x;
    const int lane = tid & 31, wave = tid >> 5;
    const int wm = wave >> 1, wn = wave & 1;             // 4 x 2 waves
    const int l16 = lane & 15, half = lane >> 4;
    const int m0 = blockIdx.x * 128, n0 = blockIdx.y * 128;

    f32x8 acc[2][4];
#pragma unroll
    for (int i = 0; i < 2; ++i)
#pragma unroll
        for (int j = 0; j < 4; ++j) {
            float bv = bias[n0 + wn * 64 + j * 16 + l16];
            acc[i][j] = (f32x8){bv, bv, bv, bv, bv, bv, bv, bv};
        }

    const int ar = tid >> 1, ac = (tid & 1) * 16;        // both tiles: 128 rows x 32
    const bf16* ga = A  + (size_t)(m0 + ar) * K + ac;
    const bf16* gb = Bt + (size_t)(n0 + ar) * K + ac;

    for (int k0 = 0; k0 < K; k0 += 32) {
        // batch all global loads into registers, then drain to LDS
        bf16x8 t0 = *(const bf16x8*)(ga + k0);
        bf16x8 t1 = *(const bf16x8*)(ga + k0 + 8);
        bf16x8 t2 = *(const bf16x8*)(gb + k0);
        bf16x8 t3 = *(const bf16x8*)(gb + k0 + 8);
        if (k0 + 32 < K) {                               // gfx1250 global_prefetch_b8 (WGP scope)
            __builtin_prefetch(ga + k0 + 32, 0, 3);
            __builtin_prefetch(gb + k0 + 32, 0, 3);
        }
        *(bf16x8*)&sA[ar][ac]     = t0;
        *(bf16x8*)&sA[ar][ac + 8] = t1;
        *(bf16x8*)&sB[ar][ac]     = t2;
        *(bf16x8*)&sB[ar][ac + 8] = t3;
        __syncthreads();

        bf16x16 af[2], bfr[4];
#pragma unroll
        for (int i = 0; i < 2; ++i) {
            int row = wm * 32 + i * 16 + l16;
            af[i] = mk16(*(const bf16x8*)&sA[row][half * 8],
                         *(const bf16x8*)&sA[row][16 + half * 8]);
        }
#pragma unroll
        for (int j = 0; j < 4; ++j) {
            int row = wn * 64 + j * 16 + l16;
            bfr[j] = mk16(*(const bf16x8*)&sB[row][half * 8],
                          *(const bf16x8*)&sB[row][16 + half * 8]);
        }
#pragma unroll
        for (int i = 0; i < 2; ++i)
#pragma unroll
            for (int j = 0; j < 4; ++j)
                acc[i][j] = __builtin_amdgcn_wmma_f32_16x16x32_bf16(
                    false, af[i], false, bfr[j], (short)0, acc[i][j], false, false);
        __syncthreads();
    }
#pragma unroll
    for (int i = 0; i < 2; ++i)
#pragma unroll
        for (int j = 0; j < 4; ++j)
#pragma unroll
            for (int v = 0; v < 8; ++v)
                C[(size_t)(m0 + wm * 32 + i * 16 + v + 8 * half) * N +
                  n0 + wn * 64 + j * 16 + l16] = acc[i][j][v];
}

// ---------------------------------------------------------------- RoPE + head layout
__global__ __launch_bounds__(256) void rope_layout_kernel(const float* __restrict__ qs,
                                                          const float* __restrict__ ks,
                                                          const float* __restrict__ vs,
                                                          bf16* __restrict__ qh,
                                                          bf16* __restrict__ kh,
                                                          bf16* __restrict__ vt) {
    int idx = blockIdx.x * 256 + threadIdx.x;   // B*S*D
    if (idx >= MM * DD) return;
    const int d = idx & 63;
    const int h = (idx >> 6) & 15;
    const int s = (idx >> 10) % SS;
    const int b = (idx >> 10) / SS;
    const size_t src = (size_t)((idx >> 10)) * DD + h * HD + d;
    const int i = d & 31;
    // inv = 10000^(-i/32)
    const float inv = exp2f(-(float)i * (13.287712379549449f / 32.0f));
    const float ang = (float)s * inv;
    const float c = cosf(ang), sn = sinf(ang);
    const float qv = qs[src], kv = ks[src];
    const size_t partner = src + ((d < 32) ? 32 : -32);
    const float qo = qs[partner], ko = ks[partner];
    const float qr = (d < 32) ? (qv * c - qo * sn) : (qv * c + qo * sn);
    const float kr = (d < 32) ? (kv * c - ko * sn) : (kv * c + ko * sn);
    const size_t hidx = ((size_t)(b * NH + h) * SS + s) * HD + d;
    qh[hidx] = (bf16)qr;
    kh[hidx] = (bf16)kr;
    vt[((size_t)(b * NH + h) * HD + d) * SS + s] = (bf16)vs[src];
}

// ---------------------------------------------------------------- flash attention (WMMA)
// block 128 = 4 waves, each wave owns a 16-row q tile; key tiles of 32 (causal).
__global__ __launch_bounds__(128) void attn_kernel(const bf16* __restrict__ qh,
                                                   const bf16* __restrict__ kh,
                                                   const bf16* __restrict__ vt,
                                                   bf16* __restrict__ ctx) {
    const int bh = blockIdx.x;                  // B*H
    const int b = bh >> 4, h = bh & 15;
    const int wave = threadIdx.x >> 5, lane = threadIdx.x & 31;
    const int l16 = lane & 15, half = lane >> 4;
    const int q0 = blockIdx.y * 64 + wave * 16;

    __shared__ __align__(16) bf16 sP[4][16][40];

    const bf16* Q = qh + (size_t)bh * SS * HD;
    const bf16* Kp = kh + (size_t)bh * SS * HD;
    const bf16* V = vt + (size_t)bh * HD * SS;

    // Q A-fragments for the two K-steps over hd=64
    bf16x16 qf[2];
#pragma unroll
    for (int ks = 0; ks < 2; ++ks) {
        const bf16* row = Q + (size_t)(q0 + l16) * HD + ks * 32;
        qf[ks] = mk16(*(const bf16x8*)(row + half * 8),
                      *(const bf16x8*)(row + 16 + half * 8));
    }

    f32x8 o[4];
#pragma unroll
    for (int c = 0; c < 4; ++c) o[c] = (f32x8){0, 0, 0, 0, 0, 0, 0, 0};
    float m2[8], lsum[8];
#pragma unroll
    for (int v = 0; v < 8; ++v) { m2[v] = -1.0e30f; lsum[v] = 0.f; }

    const float SC2 = 1.4426950408889634f / 8.0f;   // log2(e)/sqrt(hd)
    const int ntiles = (q0 + 16 + 31) >> 5;

    for (int t = 0; t < ntiles; ++t) {
        const int kb = t * 32;
        if (t + 1 < ntiles) {                        // gfx1250 global_prefetch_b8 (WGP scope)
            __builtin_prefetch(Kp + (size_t)(kb + 32 + l16) * HD, 0, 3);
            __builtin_prefetch(V + (size_t)l16 * SS + kb + 32, 0, 3);
        }
        // ---- scores: two 16x16 n-tiles, K = 64 (2 WMMA steps each)
        f32x8 sf[2];
#pragma unroll
        for (int j = 0; j < 2; ++j) {
            sf[j] = (f32x8){0, 0, 0, 0, 0, 0, 0, 0};
#pragma unroll
            for (int ks = 0; ks < 2; ++ks) {
                const bf16* krow = Kp + (size_t)(kb + j * 16 + l16) * HD + ks * 32;
                bf16x16 kf = mk16(*(const bf16x8*)(krow + half * 8),
                                  *(const bf16x8*)(krow + 16 + half * 8));
                sf[j] = __builtin_amdgcn_wmma_f32_16x16x32_bf16(
                    false, qf[ks], false, kf, (short)0, sf[j], false, false);
            }
        }
        // ---- scale + causal mask (row = v + 8*half, col = lane&15)
#pragma unroll
        for (int j = 0; j < 2; ++j) {
            const int key = kb + j * 16 + l16;
#pragma unroll
            for (int v = 0; v < 8; ++v) {
                float s = sf[j][v] * SC2;
                const int qr = q0 + v + 8 * half;
                sf[j][v] = (key > qr) ? -1.0e30f : s;
            }
        }
        // ---- online softmax (base 2), stats replicated in each 16-lane group
#pragma unroll
        for (int v = 0; v < 8; ++v) {
            float tmax = fmaxf(sf[0][v], sf[1][v]);
#pragma unroll
            for (int off = 1; off < 16; off <<= 1)
                tmax = fmaxf(tmax, __shfl_xor(tmax, off, 32));
            const float mn = fmaxf(m2[v], tmax);
            const float alpha = exp2f(m2[v] - mn);
            m2[v] = mn;
            float p0 = exp2f(sf[0][v] - mn);
            float p1 = exp2f(sf[1][v] - mn);
            sf[0][v] = p0; sf[1][v] = p1;
            float rs = p0 + p1;
#pragma unroll
            for (int off = 1; off < 16; off <<= 1)
                rs += __shfl_xor(rs, off, 32);
            lsum[v] = lsum[v] * alpha + rs;
#pragma unroll
            for (int c = 0; c < 4; ++c) o[c][v] *= alpha;
        }
        // ---- probs -> LDS -> A fragment
#pragma unroll
        for (int j = 0; j < 2; ++j)
#pragma unroll
            for (int v = 0; v < 8; ++v)
                sP[wave][v + 8 * half][j * 16 + l16] = (bf16)sf[j][v];
        asm volatile("s_wait_dscnt 0" ::: "memory");
        bf16x16 pf = mk16(*(const bf16x8*)&sP[wave][l16][half * 8],
                          *(const bf16x8*)&sP[wave][l16][16 + half * 8]);
        // ---- out += P(16x32) @ V(32x64): 4 hd chunks
#pragma unroll
        for (int c = 0; c < 4; ++c) {
            const bf16* vrow = V + (size_t)(c * 16 + l16) * SS + kb;
            bf16x16 vf = mk16(*(const bf16x8*)(vrow + half * 8),
                              *(const bf16x8*)(vrow + 16 + half * 8));
            o[c] = __builtin_amdgcn_wmma_f32_16x16x32_bf16(
                false, pf, false, vf, (short)0, o[c], false, false);
        }
    }
    // ---- normalize + write ctx (B*S, 1024) bf16
#pragma unroll
    for (int c = 0; c < 4; ++c)
#pragma unroll
        for (int v = 0; v < 8; ++v) {
            float val = o[c][v] / lsum[v];
            ctx[(size_t)(b * SS + q0 + v + 8 * half) * DD +
                h * HD + c * 16 + l16] = (bf16)val;
        }
}

// ---------------------------------------------------------------- launch
extern "C" void kernel_launch(void* const* d_in, const int* in_sizes, int n_in,
                              void* d_out, int out_size, void* d_ws, size_t ws_size,
                              hipStream_t stream) {
    const float* x     = (const float*)d_in[0];
    const float* Wq    = (const float*)d_in[1];
    const float* bq    = (const float*)d_in[2];
    const float* Wo    = (const float*)d_in[3];
    const float* bo    = (const float*)d_in[4];
    const float* Wk    = (const float*)d_in[5];
    const float* bk    = (const float*)d_in[6];
    const float* k_w1  = (const float*)d_in[7];
    const float* k_b1  = (const float*)d_in[8];
    const float* k_w2  = (const float*)d_in[9];
    const float* k_b2  = (const float*)d_in[10];
    const float* k_gAw = (const float*)d_in[11];
    const float* k_gAb = (const float*)d_in[12];
    const float* k_gBw = (const float*)d_in[13];
    const float* k_gBb = (const float*)d_in[14];
    const float* Wv    = (const float*)d_in[15];
    const float* bv    = (const float*)d_in[16];
    const float* v_w1  = (const float*)d_in[17];
    const float* v_b1  = (const float*)d_in[18];
    const float* v_w2  = (const float*)d_in[19];
    const float* v_b2  = (const float*)d_in[20];
    const float* v_gAw = (const float*)d_in[21];
    const float* v_gAb = (const float*)d_in[22];
    const float* v_gBw = (const float*)d_in[23];
    const float* v_gBb = (const float*)d_in[24];
    float* out = (float*)d_out;

    char* ws = (char*)d_ws;
    const size_t MB = 1u << 20;
    bf16* xb    = (bf16*)(ws + 0 * MB);       // 8 MB
    bf16* Wtq   = (bf16*)(ws + 8 * MB);       // 2 MB each
    bf16* Wtk   = (bf16*)(ws + 10 * MB);
    bf16* Wtv   = (bf16*)(ws + 12 * MB);
    bf16* Wto   = (bf16*)(ws + 14 * MB);
    float* qs   = (float*)(ws + 16 * MB);     // 16 MB each
    float* kst  = (float*)(ws + 32 * MB);
    float* vst  = (float*)(ws + 48 * MB);
    bf16* qh    = (bf16*)(ws + 64 * MB);      // 8 MB each
    bf16* kh    = (bf16*)(ws + 72 * MB);
    bf16* vt    = (bf16*)(ws + 80 * MB);
    bf16* ctxb  = (bf16*)(ws + 88 * MB);      // 8 MB
    char* sm    = ws + 96 * MB;               // small buffers
    float* z    = (float*)(sm + 0x00000);
    float* hK   = (float*)(sm + 0x10000);
    float* hV   = (float*)(sm + 0x20000);
    float* cK   = (float*)(sm + 0x30000);
    float* cV   = (float*)(sm + 0x40000);
    float* Ak   = (float*)(sm + 0x50000);     // 64 KB each
    float* Bk   = (float*)(sm + 0x60000);
    float* Av   = (float*)(sm + 0x70000);
    float* Bv   = (float*)(sm + 0x80000);
    float* xaK  = (float*)(sm + 0x90000);     // 128 KB each
    float* xaV  = (float*)(sm + 0xB0000);

    const int NXD = MM * DD;                  // 4M elems
    // 1) converts
    cvt_bf16_kernel<<<(NXD + 255) / 256, 256, 0, stream>>>(x, xb, NXD);
    wtrans_kernel<<<dim3(32, 32), 256, 0, stream>>>(Wq, Wtq);
    wtrans_kernel<<<dim3(32, 32), 256, 0, stream>>>(Wk, Wtk);
    wtrans_kernel<<<dim3(32, 32), 256, 0, stream>>>(Wv, Wtv);
    wtrans_kernel<<<dim3(32, 32), 256, 0, stream>>>(Wo, Wto);
    // 2) hyper-net
    mean_kernel<<<(NB * DD + 255) / 256, 256, 0, stream>>>(x, z);
    dense_small_kernel<<<(NB * HID + 255) / 256, 256, 0, stream>>>(z, k_w1, k_b1, hK, DD, HID, 1);
    dense_small_kernel<<<(NB * HID + 255) / 256, 256, 0, stream>>>(z, v_w1, v_b1, hV, DD, HID, 1);
    dense_small_kernel<<<(NB * DYN + 255) / 256, 256, 0, stream>>>(hK, k_w2, k_b2, cK, HID, DYN, 0);
    dense_small_kernel<<<(NB * DYN + 255) / 256, 256, 0, stream>>>(hV, v_w2, v_b2, cV, HID, DYN, 0);
    dense_small_kernel<<<(NB * RK * DD + 255) / 256, 256, 0, stream>>>(cK, k_gAw, k_gAb, Ak, DYN, RK * DD, 0);
    dense_small_kernel<<<(NB * RK * DD + 255) / 256, 256, 0, stream>>>(cK, k_gBw, k_gBb, Bk, DYN, DD * RK, 0);
    dense_small_kernel<<<(NB * RK * DD + 255) / 256, 256, 0, stream>>>(cV, v_gAw, v_gAb, Av, DYN, RK * DD, 0);
    dense_small_kernel<<<(NB * RK * DD + 255) / 256, 256, 0, stream>>>(cV, v_gBw, v_gBb, Bv, DYN, DD * RK, 0);
    // 3) big projections (WMMA)
    gemm_bf16_kernel<<<dim3(MM / 128, DD / 128), 256, 0, stream>>>(xb, Wtq, bq, qs, MM, DD, DD);
    gemm_bf16_kernel<<<dim3(MM / 128, DD / 128), 256, 0, stream>>>(xb, Wtk, bk, kst, MM, DD, DD);
    gemm_bf16_kernel<<<dim3(MM / 128, DD / 128), 256, 0, stream>>>(xb, Wtv, bv, vst, MM, DD, DD);
    // 4) low-rank dynamic corrections
    xa_kernel<<<MM, 256, 0, stream>>>(x, Ak, xaK);
    xa_kernel<<<MM, 256, 0, stream>>>(x, Av, xaV);
    add_dyn_kernel<<<(NXD + 255) / 256, 256, 0, stream>>>(kst, xaK, Bk);
    add_dyn_kernel<<<(NXD + 255) / 256, 256, 0, stream>>>(vst, xaV, Bv);
    // 5) rope + head layout
    rope_layout_kernel<<<(NXD + 255) / 256, 256, 0, stream>>>(qs, kst, vst, qh, kh, vt);
    // 6) flash attention (WMMA)
    attn_kernel<<<dim3(NB * NH, SS / 64), 128, 0, stream>>>(qh, kh, vt, ctxb);
    // 7) output projection (WMMA)
    gemm_bf16_kernel<<<dim3(MM / 128, DD / 128), 256, 0, stream>>>(ctxb, Wto, bo, out, MM, DD, DD);
    (void)in_sizes; (void)n_in; (void)out_size; (void)ws_size;
}
